// KVMemory_40630390621011
// MI455X (gfx1250) — matrix operand order repaired and checked by reading, not compiled, not run on hardware
//
#include <hip/hip_runtime.h>

typedef __attribute__((ext_vector_type(16))) __bf16 v16bf;
typedef __attribute__((ext_vector_type(8)))  float  v8f;
typedef __attribute__((ext_vector_type(8)))  __bf16 bf16x8;

#define KF      512
#define NQ      2048
#define MEM     131072
#define TOPK    32
#define QB      16
#define NQB     (NQ/QB)        // 128 query blocks
#define NSLICE  32
#define SLICE   (MEM/NSLICE)   // 4096 keys per slice
#define CHUNK   128
#define NCHUNK  (SLICE/CHUNK)  // 32 chunks per slice
#define PART_ELEMS ((size_t)NQB*NSLICE*QB*TOPK)  // 2,097,152

// Wave-register top-32: one sorted (score,idx) entry per lane, descending,
// lane 31 holds the current minimum. Insert any lane-candidates above min.
__device__ __forceinline__ void topk32_insert(float v, int idx,
                                              float& ls, int& li, int lane) {
  float minv = __shfl(ls, 31, 32);
  unsigned mask = (unsigned)__ballot(v > minv);
  while (mask) {
    int src = __ffs(mask) - 1;
    mask &= mask - 1;
    float cv = __shfl(v, src, 32);
    int   ci = __shfl(idx, src, 32);
    int  pos = __popc((unsigned)__ballot(ls >= cv));  // insertion position
    float ps = __shfl_up(ls, 1, 32);
    int   pi = __shfl_up(li, 1, 32);
    if (lane > pos)       { ls = ps; li = pi; }
    else if (lane == pos) { ls = cv; li = ci; }
  }
}

__global__ __launch_bounds__(256) void score_topk_kernel(
    const float* __restrict__ q, const float* __restrict__ kmem,
    float* __restrict__ part_s, int* __restrict__ part_i) {
  __shared__ __bf16 sQh[QB * KF];      // 16 KB  Q hi
  __shared__ __bf16 sQl[QB * KF];      // 16 KB  Q lo (split residual)
  __shared__ __bf16 sKh[KF * CHUNK];   // 128 KB K hi, [key][feat], not transposed
  __shared__ __bf16 sKl[KF * CHUNK];   // 128 KB K lo, [key][feat]
  __shared__ float  sS [QB * CHUNK];   // 8 KB   score tile

  const int tid   = threadIdx.x;
  const int lane  = tid & 31;
  const int wv    = tid >> 5;          // 8 waves, wave = N-tile
  const int qb    = blockIdx.x;
  const int slice = blockIdx.y;

  // ---- stage Q block, split f32 -> bf16 hi + bf16 lo ----
  {
    const float4* qs = (const float4*)(q + (size_t)qb * QB * KF);
    #pragma unroll
    for (int i = 0; i < (QB*KF/4)/256; ++i) {      // 8 iters
      int f4 = tid + i * 256;
      float4 x = qs[f4];
      float vals[4] = {x.x, x.y, x.z, x.w};
      int base = f4 * 4;
      #pragma unroll
      for (int j = 0; j < 4; ++j) {
        __bf16 h = (__bf16)vals[j];
        sQh[base + j] = h;
        sQl[base + j] = (__bf16)(vals[j] - (float)h);
      }
    }
  }

  float ls[2]; int li[2];
  ls[0] = ls[1] = -__builtin_inff();
  li[0] = li[1] = 0;

  const int m    = lane & 15;                 // A row for this lane
  const int koff = (lane < 16) ? 0 : 8;       // bf16 A K-halves per lane group

  for (int c = 0; c < NCHUNK; ++c) {
    const size_t key0 = (size_t)slice * SLICE + (size_t)c * CHUNK;
    __syncthreads();  // prior selection done before LDS overwrite
    // ---- stream CHUNK keys, split into LDS [key][feat] ----
    {
      const float4* ks = (const float4*)(kmem + key0 * KF);
      #pragma unroll 4
      for (int i = 0; i < (CHUNK*KF/4)/256; ++i) { // 64 iters
        int f4  = tid + i * 256;
        int key = f4 >> 7;            // 128 float4 per key row
        int fo  = (f4 & 127) * 4;
        float4 x = ks[f4];
        float vals[4] = {x.x, x.y, x.z, x.w};
        #pragma unroll
        for (int j = 0; j < 4; ++j) {
          __bf16 h = (__bf16)vals[j];
          sKh[key * KF + fo + j] = h;
          sKl[key * KF + fo + j] = (__bf16)(vals[j] - (float)h);
        }
      }
    }
    __syncthreads();

    // ---- overlap: prefetch next chunk's 256 KB into cache during compute ----
    if (c + 1 < NCHUNK) {
      const char* nxt = (const char*)(kmem + (key0 + CHUNK) * KF);
      #pragma unroll
      for (int i = 0; i < (CHUNK*KF*4)/(128*256); ++i)  // 8 cachelines/thread
        __builtin_prefetch(nxt + ((size_t)(tid + i * 256)) * 128, 0, 1);
    }

    // ---- WMMA: one 16x16 tile per wave, K=512 in steps of 32 ----
    v8f acc = {};
    for (int kk = 0; kk < KF; kk += 32) {
      bf16x8 a0 = *(const bf16x8*)&sQh[m * KF + kk + koff];
      bf16x8 a1 = *(const bf16x8*)&sQh[m * KF + kk + 16 + koff];
      bf16x8 a2 = *(const bf16x8*)&sQl[m * KF + kk + koff];
      bf16x8 a3 = *(const bf16x8*)&sQl[m * KF + kk + 16 + koff];
      v16bf ah = __builtin_shufflevector(a0, a1, 0,1,2,3,4,5,6,7,8,9,10,11,12,13,14,15);
      v16bf al = __builtin_shufflevector(a2, a3, 0,1,2,3,4,5,6,7,8,9,10,11,12,13,14,15);

      int brow = (wv * 16 + (lane & 15)) * KF + kk + koff;   // B lane = N column, K halves like A
      bf16x8 b0 = *(const bf16x8*)&sKh[brow];
      bf16x8 b1 = *(const bf16x8*)&sKh[brow + 16];
      bf16x8 b2 = *(const bf16x8*)&sKl[brow];
      bf16x8 b3 = *(const bf16x8*)&sKl[brow + 16];
      v16bf bh = __builtin_shufflevector(b0, b1, 0,1,2,3,4,5,6,7,8,9,10,11,12,13,14,15);
      v16bf bl = __builtin_shufflevector(b2, b3, 0,1,2,3,4,5,6,7,8,9,10,11,12,13,14,15);

      // bf16x2 split: hi*hi + hi*lo + lo*hi  (lo*lo negligible)
      acc = __builtin_amdgcn_wmma_f32_16x16x32_bf16(false, ah, false, bh, (short)0, acc, false, false);
      acc = __builtin_amdgcn_wmma_f32_16x16x32_bf16(false, ah, false, bl, (short)0, acc, false, false);
      acc = __builtin_amdgcn_wmma_f32_16x16x32_bf16(false, al, false, bh, (short)0, acc, false, false);
    }

    // ---- scatter D tile to LDS scores (VGPR v: M=v+8*(lane>=16), N=lane&15)
    {
      int n     = wv * 16 + (lane & 15);
      int mbase = (lane >> 4) * 8;
      #pragma unroll
      for (int vv = 0; vv < 8; ++vv)
        sS[(mbase + vv) * CHUNK + n] = acc[vv];
    }
    __syncthreads();

    // ---- top-32 update: wave wv owns query rows 2wv, 2wv+1 ----
    #pragma unroll
    for (int rr = 0; rr < 2; ++rr) {
      int row = wv * 2 + rr;
      #pragma unroll
      for (int g = 0; g < CHUNK / 32; ++g) {
        int col = g * 32 + lane;
        float v = sS[row * CHUNK + col];
        int  gi = slice * SLICE + c * CHUNK + col;
        topk32_insert(v, gi, ls[rr], li[rr], lane);
      }
    }
  }

  // ---- emit per-slice partial top-32 lists ----
  #pragma unroll
  for (int rr = 0; rr < 2; ++rr) {
    int row = wv * 2 + rr;
    size_t base = ((((size_t)qb * NSLICE + slice) * QB) + row) * TOPK;
    part_s[base + lane] = ls[rr];
    part_i[base + lane] = li[rr];
  }
}

__global__ __launch_bounds__(32) void merge_topk_kernel(
    const float* __restrict__ part_s, const int* __restrict__ part_i,
    int* __restrict__ final_i) {
  const int qid  = blockIdx.x;       // 0..2047
  const int lane = threadIdx.x;      // 0..31
  const int qb = qid >> 4, row = qid & 15;
  float ls = -__builtin_inff(); int li = 0;
  for (int s = 0; s < NSLICE; ++s) {
    size_t base = ((((size_t)qb * NSLICE + s) * QB) + row) * TOPK;
    topk32_insert(part_s[base + lane], part_i[base + lane], ls, li, lane);
  }
  final_i[(size_t)qid * TOPK + lane] = li;   // descending order, like lax.top_k
}

__global__ __launch_bounds__(128) void gather_kernel(
    const float* __restrict__ kmem, const float* __restrict__ vmem,
    const int* __restrict__ final_i, float* __restrict__ out) {
  const int row = blockIdx.x;        // 0..65535  (query*32 + item)
  const int t   = threadIdx.x;       // 0..127 float4s per 512-f32 row
  const size_t idx = (size_t)final_i[row];
  const float4* ks = (const float4*)(kmem + idx * KF);
  const float4* vs = (const float4*)(vmem + idx * KF);
  float4* kd = (float4*)(out + (size_t)row * KF);
  float4* vd = (float4*)(out + (size_t)NQ * TOPK * KF + (size_t)row * KF);
  kd[t] = ks[t];
  vd[t] = vs[t];
}

extern "C" void kernel_launch(void* const* d_in, const int* in_sizes, int n_in,
                              void* d_out, int out_size, void* d_ws, size_t ws_size,
                              hipStream_t stream) {
  const float* q    = (const float*)d_in[0];
  const float* kmem = (const float*)d_in[1];
  const float* vmem = (const float*)d_in[2];
  // d_in[3] = items_per_query (compile-time 32 here)
  float* out = (float*)d_out;

  float* part_s  = (float*)d_ws;
  int*   part_i  = (int*)(part_s + PART_ELEMS);
  int*   final_i = part_i + PART_ELEMS;

  dim3 g1(NQB, NSLICE);
  score_topk_kernel<<<g1, 256, 0, stream>>>(q, kmem, part_s, part_i);
  merge_topk_kernel<<<NQ, 32, 0, stream>>>(part_s, part_i, final_i);
  gather_kernel<<<NQ * TOPK, 128, 0, stream>>>(kmem, vmem, final_i, out);
}
